// DeltaNet_31877247271579
// MI455X (gfx1250) — matrix operand order repaired and checked by reading, not compiled, not run on hardware
//
#include <hip/hip_runtime.h>

// ---------------------------------------------------------------------------
// Problem constants (match the reference)
// ---------------------------------------------------------------------------
#define B_      2
#define L_      4096
#define D_      1024
#define H_      4
#define DK_     256
#define DV_     256
#define CHUNK_  32
#define NCHUNK_ (L_ / CHUNK_)
#define ROWS_   (B_ * L_)
#define RIN_    (D_ + H_ * 6)   // 1048
#define RINP_   1056            // padded to a multiple of 32 for the GEMM
#define RHID_   (2 * D_)        // 2048
#define EPS_ID_ 0.06f
#define R_EPS_  0.025f

#define USE_ASYNC_COPY 1        // CDNA5 global_load_async_to_lds_b128 staging

// ---------------------------------------------------------------------------
// CDNA5 WMMA plumbing (wave32, 16x16x32 bf16 -> f32)
// ---------------------------------------------------------------------------
typedef __attribute__((ext_vector_type(16))) __bf16 v16bf;
typedef __attribute__((ext_vector_type(8)))  float  v8f;

union BFrag16 { v16bf v; unsigned short h[16]; uint4 q[2]; };

__device__ inline unsigned short f2bf(float f) {
  unsigned int u = __float_as_uint(f);
  return (unsigned short)((u + 0x7FFFu + ((u >> 16) & 1u)) >> 16);  // RNE
}
__device__ inline float bf2f(unsigned short h) {
  return __uint_as_float(((unsigned int)h) << 16);
}
__device__ inline v8f wmma_bf16(v16bf a, v16bf b, v8f c) {
  return __builtin_amdgcn_wmma_f32_16x16x32_bf16(false, a, false, b, (short)0, c,
                                                 false, false);
}

// A fragment (16x32 bf16, row-major [m][k], ld in elements, 16B-aligned rows):
// lane element map collapses to two contiguous 16-byte runs per lane.
__device__ inline v16bf load_a_vec(const unsigned short* p, int ld) {
  int lane = threadIdx.x & 31;
  int m = lane & 15, hi = (lane >> 4) * 8;
  const unsigned short* q = p + m * ld + hi;
  BFrag16 f;
  f.q[0] = *(const uint4*)(q);
  f.q[1] = *(const uint4*)(q + 16);
  return f.v;
}
// B fragment from a K-contiguous (transposed) source: element B[k][n] = p[n*ld+k];
// per lane this is two 16-byte runs (k = hi..hi+7 and hi+16..hi+23), matching A.
__device__ inline v16bf load_bt_vec(const unsigned short* p, int ld) {
  int lane = threadIdx.x & 31;
  int n = lane & 15, hi = (lane >> 4) * 8;
  const unsigned short* q = p + n * ld + hi;
  BFrag16 f;
  f.q[0] = *(const uint4*)(q);
  f.q[1] = *(const uint4*)(q + 16);
  return f.v;
}
// C/D fragment (f32 16x16): VGPR r -> M = r + 8*hi, N = lane&15.
__device__ inline v8f load_c_f32(const float* p, int ld) {
  int lane = threadIdx.x & 31;
  int n = lane & 15, hi = (lane >> 4) * 8;
  v8f c;
#pragma unroll
  for (int r = 0; r < 8; ++r) c[r] = p[(r + hi) * ld + n];
  return c;
}
__device__ inline void store_c_f32(float* p, int ld, v8f c) {
  int lane = threadIdx.x & 31;
  int n = lane & 15, hi = (lane >> 4) * 8;
#pragma unroll
  for (int r = 0; r < 8; ++r) p[(r + hi) * ld + n] = c[r];
}

// 16-byte global -> LDS copy (async TDM-counter path on CDNA5).
__device__ inline void copy16(unsigned short* lds, const unsigned short* g) {
#if USE_ASYNC_COPY
  unsigned laddr = (unsigned)(unsigned long long)(void*)lds;  // low 32b = LDS offset
  asm volatile("global_load_async_to_lds_b128 %0, %1, off"
               :: "v"(laddr), "v"(g) : "memory");
#else
  *(uint4*)lds = *(const uint4*)g;
#endif
}
__device__ inline void wait_async() {
#if USE_ASYNC_COPY
#if __has_builtin(__builtin_amdgcn_s_wait_asynccnt)
  __builtin_amdgcn_s_wait_asynccnt(0);
#else
  asm volatile("s_wait_asynccnt 0x0" ::: "memory");
#endif
#endif
}

// Block-wide (256-thread) sum reduction; returns result to all threads.
__device__ inline float block_reduce256(float v, float* red) {
  int t = threadIdx.x;
  red[t] = v;
  __syncthreads();
  for (int s = 128; s > 0; s >>= 1) {
    if (t < s) red[t] += red[t + s];
    __syncthreads();
  }
  float r = red[0];
  __syncthreads();
  return r;
}

// ---------------------------------------------------------------------------
// Conversions: f32 -> bf16 (flat) and f32 [K][N] -> bf16 transposed [N][Kpad]
// ---------------------------------------------------------------------------
__global__ __launch_bounds__(256) void cvt_bf16_kernel(
    const float* __restrict__ A, unsigned short* __restrict__ O, long n) {
  long i = (long)blockIdx.x * 256 + threadIdx.x;
  if (i < n) O[i] = f2bf(A[i]);
}
__global__ __launch_bounds__(256) void transpose_bf16_kernel(
    const float* __restrict__ W, unsigned short* __restrict__ WT, int K, int N,
    int Kpad) {
  long i = (long)blockIdx.x * 256 + threadIdx.x;
  if (i >= (long)N * Kpad) return;
  int n = (int)(i / Kpad), k = (int)(i % Kpad);
  WT[i] = (k < K) ? f2bf(W[(size_t)k * N + n]) : (unsigned short)0;
}

// ---------------------------------------------------------------------------
// Kernel: bf16-WMMA GEMM  C[M,N] = act(A[M,K] @ B[K,N] + bias)
// A bf16 [M][K]; B given transposed bf16 [N][K]. M,N multiples of 64; K of 32.
// Double-buffered LDS staging via async b128 copies (one per thread per tile).
// ---------------------------------------------------------------------------
#define ACT_NONE 0
#define ACT_GELU 1

__global__ __launch_bounds__(256) void gemm_bf16_kernel(
    const unsigned short* __restrict__ A, const unsigned short* __restrict__ BT,
    const float* __restrict__ bias, float* __restrict__ C, int M, int N, int K,
    int act) {
  __shared__ __align__(16) unsigned short As[2][64 * 32];
  __shared__ __align__(16) unsigned short Bs[2][64 * 32];
  const int t = threadIdx.x, w = t >> 5, lane = t & 31;
  const int bm = blockIdx.y * 64, bn = blockIdx.x * 64;
  const int tm = w & 3, tn = w >> 2;
  const int sr = t >> 2, sq = (t & 3) * 8;  // staging: row, 8-elem col offset
  v8f c0 = {}, c1 = {};

  copy16(&As[0][sr * 32 + sq], A + (size_t)(bm + sr) * K + sq);
  copy16(&Bs[0][sr * 32 + sq], BT + (size_t)(bn + sr) * K + sq);
  wait_async();
  __syncthreads();

  int buf = 0;
  for (int k0 = 0; k0 < K; k0 += 32) {
    if (k0 + 32 < K) {
      copy16(&As[buf ^ 1][sr * 32 + sq], A + (size_t)(bm + sr) * K + k0 + 32 + sq);
      copy16(&Bs[buf ^ 1][sr * 32 + sq], BT + (size_t)(bn + sr) * K + k0 + 32 + sq);
    }
    v16bf a  = load_a_vec(&As[buf][16 * tm * 32], 32);
    v16bf b0 = load_bt_vec(&Bs[buf][(32 * tn) * 32], 32);
    v16bf b1 = load_bt_vec(&Bs[buf][(32 * tn + 16) * 32], 32);
    c0 = wmma_bf16(a, b0, c0);
    c1 = wmma_bf16(a, b1, c1);
    wait_async();
    __syncthreads();
    buf ^= 1;
  }

  const int n = lane & 15, hi8 = (lane >> 4) * 8;
#pragma unroll
  for (int half = 0; half < 2; ++half) {
    v8f cc = half ? c1 : c0;
    int gn = bn + 32 * tn + 16 * half + n;
    float bv = bias ? bias[gn] : 0.f;
#pragma unroll
    for (int r = 0; r < 8; ++r) {
      int gm = bm + 16 * tm + r + hi8;
      float v = cc[r] + bv;
      if (act == ACT_GELU) v = 0.5f * v * (1.f + erff(v * 0.70710678118f));
      C[(size_t)gm * N + gn] = v;
    }
  }
}

// ---------------------------------------------------------------------------
// Causal depthwise conv (+ optional SiLU). X,Y: [B,L,C]; W: [C,Kw].
// ---------------------------------------------------------------------------
__global__ __launch_bounds__(256) void dwconv_kernel(
    const float* __restrict__ X, const float* __restrict__ Wc,
    float* __restrict__ Y, int Kw, int do_silu) {
  size_t idx = (size_t)blockIdx.x * blockDim.x + threadIdx.x;
  const size_t total = (size_t)ROWS_ * D_;
  if (idx >= total) return;
  int c = (int)(idx % D_);
  size_t bl = idx / D_;
  int l = (int)(bl % L_);
  size_t bb = bl / L_;
  float acc = 0.f;
  for (int j = 0; j < Kw; ++j) {
    int ls = l - (Kw - 1) + j;
    if (ls >= 0) acc += X[((size_t)bb * L_ + ls) * D_ + c] * Wc[c * Kw + j];
  }
  if (do_silu) acc = acc / (1.f + expf(-acc));
  Y[idx] = acc;
}

// ---------------------------------------------------------------------------
// beta = sigmoid(x@Wb); id_scale = EPS + sig(alpha)*sig(x@Wid+bid)
// ---------------------------------------------------------------------------
__global__ __launch_bounds__(256) void beta_id_kernel(
    const float* __restrict__ x, const float* __restrict__ Wb,
    const float* __restrict__ Wid, const float* __restrict__ bid,
    const float* __restrict__ alpha_id, float* __restrict__ beta,
    float* __restrict__ idsc) {
  __shared__ float red[256];
  const int row = blockIdx.x, t = threadIdx.x;
  const float* xr = x + (size_t)row * D_;
  float pb[H_] = {0, 0, 0, 0}, pi[H_] = {0, 0, 0, 0};
  for (int jj = 0; jj < 4; ++jj) {
    int ix = t + 256 * jj;
    float xv = xr[ix];
#pragma unroll
    for (int h = 0; h < H_; ++h) {
      pb[h] += xv * Wb[ix * H_ + h];
      pi[h] += xv * Wid[ix * H_ + h];
    }
  }
  for (int h = 0; h < H_; ++h) {
    float db = block_reduce256(pb[h], red);
    float di = block_reduce256(pi[h], red);
    if (t == 0) {
      beta[row * H_ + h] = 1.f / (1.f + expf(-db));
      float g = 1.f / (1.f + expf(-(di + bid[h])));
      float a = 1.f / (1.f + expf(-alpha_id[h]));
      idsc[row * H_ + h] = EPS_ID_ + a * g;
    }
    __syncthreads();
  }
}

// ---------------------------------------------------------------------------
// l2norm(q), l2norm(k), kb=k*beta, vb=v*beta; emit bf16 operand layouts:
//   row-major [b,h,l,d]:  qn, kn, kbn          (A / k^T-B fragments)
//   transposed [b,h,d,l]: kTn, kbTn, vbTn      (K-contiguous B / A fragments)
// ---------------------------------------------------------------------------
__global__ __launch_bounds__(256) void prep_kernel(
    const float* __restrict__ qs, const float* __restrict__ ks,
    const float* __restrict__ vs, const float* __restrict__ beta,
    unsigned short* __restrict__ qn, unsigned short* __restrict__ kn,
    unsigned short* __restrict__ kTn, unsigned short* __restrict__ kbn,
    unsigned short* __restrict__ kbTn, unsigned short* __restrict__ vbTn) {
  __shared__ float red[256];
  const int row = blockIdx.x, t = threadIdx.x;
  const int b = row / L_, l = row % L_;
  for (int h = 0; h < H_; ++h) {
    int c = h * DK_ + t;
    float qv = qs[(size_t)row * D_ + c];
    float kv = ks[(size_t)row * D_ + c];
    float vv = vs[(size_t)row * D_ + c];
    float qss = block_reduce256(qv * qv, red);
    float kss = block_reduce256(kv * kv, red);
    float bet = beta[row * H_ + h];
    int bh = b * H_ + h;
    size_t o  = ((size_t)bh * L_ + l) * DK_ + t;
    size_t oT = ((size_t)bh * DK_ + t) * L_ + l;
    float qnv = qv * rsqrtf(qss + 1e-6f);
    float knv = kv * rsqrtf(kss + 1e-6f);
    qn[o] = f2bf(qnv);
    kn[o] = f2bf(knv);
    kbn[o] = f2bf(knv * bet);
    kTn[oT] = f2bf(knv);
    kbTn[oT] = f2bf(knv * bet);
    vbTn[oT] = f2bf(vv * bet);
  }
}

// ---------------------------------------------------------------------------
// Chunkwise delta rule. One block (8 waves) per (b,h). S: f32 in global
// (L2-resident) + bf16 transposed mirror ST for vectorized B fragments.
// u2 kept transposed in LDS so all its fragment loads are contiguous.
// ---------------------------------------------------------------------------
__global__ __launch_bounds__(256) void delta_kernel(
    const unsigned short* __restrict__ qn, const unsigned short* __restrict__ kn,
    const unsigned short* __restrict__ kTn,
    const unsigned short* __restrict__ kbn,
    const unsigned short* __restrict__ kbTn,
    const unsigned short* __restrict__ vbTn, float* __restrict__ Sg,
    unsigned short* __restrict__ STg, float* __restrict__ delta) {
  __shared__ __align__(16) float Tf[32 * 32];
  __shared__ __align__(16) unsigned short Tb[32 * 32];    // T bf16 (A frags)
  __shared__ __align__(16) unsigned short ab[32 * 32];    // attn bf16 (A frags)
  __shared__ __align__(16) unsigned short ubT[256 * 32];  // u/u2 transposed [dv][m]
  __shared__ __align__(16) unsigned short wb[32 * 256];   // -w row-major [m][dk]

  const int bh = blockIdx.x;
  const int b = bh / H_, h = bh % H_;
  const int t = threadIdx.x, w = t >> 5, lane = t & 31;
  const int cn = lane & 15, hi8 = (lane >> 4) * 8;

  const unsigned short* qg   = qn   + (size_t)bh * L_ * DK_;
  const unsigned short* kg   = kn   + (size_t)bh * L_ * DK_;
  const unsigned short* kTg  = kTn  + (size_t)bh * DK_ * L_;
  const unsigned short* kbg  = kbn  + (size_t)bh * L_ * DK_;
  const unsigned short* kbTg = kbTn + (size_t)bh * DK_ * L_;
  const unsigned short* vbTg = vbTn + (size_t)bh * DV_ * L_;
  float* S = Sg + (size_t)bh * DK_ * DV_;
  unsigned short* ST = STg + (size_t)bh * DV_ * DK_;  // ST[dv][dk] bf16

  for (int i = t; i < DK_ * DV_; i += 256) { S[i] = 0.f; ST[i] = 0; }
  __syncthreads();

  for (int c = 0; c < NCHUNK_; ++c) {
    const int l0 = c * CHUNK_;
    if (l0 + CHUNK_ < L_)
      __builtin_prefetch(qg + (size_t)(l0 + CHUNK_) * DK_ + lane * 8, 0, 1);

    // Phase 1: T = strict_lower( -(kb @ k^T) ), K=256
    if (w < 4) {
      int tm = w >> 1, tn = w & 1;
      v8f acc = {};
      for (int kk = 0; kk < DK_; kk += 32) {
        v16bf a  = load_a_vec(kbg + (size_t)(l0 + 16 * tm) * DK_ + kk, DK_);
        v16bf bB = load_bt_vec(kg + (size_t)(l0 + 16 * tn) * DK_ + kk, DK_);
        acc = wmma_bf16(a, bB, acc);
      }
#pragma unroll
      for (int r = 0; r < 8; ++r) {
        int m = 16 * tm + r + hi8, nn = 16 * tn + cn;
        Tf[m * 32 + nn] = (nn < m) ? -acc[r] : 0.f;
      }
    }
    __syncthreads();

    // Phase 2: serial forward substitution  T[i,:] += T[i,:] @ T
    for (int i = 1; i < 32; ++i) {
      float add = 0.f;
      if (t < 32)
        for (int j = 0; j < i; ++j) add += Tf[i * 32 + j] * Tf[j * 32 + t];
      __syncthreads();
      if (t < 32) Tf[i * 32 + t] += add;
      __syncthreads();
    }
    for (int i = t; i < 32 * 32; i += 256)
      Tb[i] = f2bf(Tf[i] + (((i >> 5) == (i & 31)) ? 1.f : 0.f));  // T += I
    __syncthreads();

    // Phase 3: u = T@vb -> ubT (transposed);  wb = -(T@kb) row-major
    for (int it = 0; it < 4; ++it) {
      int tile = w * 4 + it, tm = tile >> 4, tj = tile & 15;
      v16bf a = load_a_vec(Tb + 16 * tm * 32, 32);
      v8f au = {}, aw = {};
      au = wmma_bf16(a, load_bt_vec(vbTg + (size_t)(16 * tj) * L_ + l0, L_), au);
      aw = wmma_bf16(a, load_bt_vec(kbTg + (size_t)(16 * tj) * L_ + l0, L_), aw);
      uint4 pk;
      unsigned short* ps = (unsigned short*)&pk;
#pragma unroll
      for (int r = 0; r < 8; ++r) {
        ps[r] = f2bf(au[r]);
        wb[(16 * tm + r + hi8) * 256 + 16 * tj + cn] = f2bf(-aw[r]);
      }
      *(uint4*)(ubT + (size_t)(16 * tj + cn) * 32 + 16 * tm + hi8) = pk;
    }
    __syncthreads();

    // Phase 4: u2 = u + (-w) @ S  -> ubT (in place, tile-owned)
    for (int it = 0; it < 4; ++it) {
      int tile = w * 4 + it, tm = tile >> 4, tj = tile & 15;
      uint4 upk = *(const uint4*)(ubT + (size_t)(16 * tj + cn) * 32 + 16 * tm + hi8);
      const unsigned short* us = (const unsigned short*)&upk;
      v8f acc;
#pragma unroll
      for (int r = 0; r < 8; ++r) acc[r] = bf2f(us[r]);
      for (int kk = 0; kk < DK_; kk += 32) {
        v16bf a  = load_a_vec(wb + 16 * tm * 256 + kk, 256);
        v16bf bS = load_bt_vec(ST + (size_t)(16 * tj) * DK_ + kk, DK_);
        acc = wmma_bf16(a, bS, acc);
      }
      uint4 pk;
      unsigned short* ps = (unsigned short*)&pk;
#pragma unroll
      for (int r = 0; r < 8; ++r) ps[r] = f2bf(acc[r]);
      *(uint4*)(ubT + (size_t)(16 * tj + cn) * 32 + 16 * tm + hi8) = pk;
    }
    __syncthreads();

    // Phase 5: attn = tril(q @ k^T) incl diag
    if (w < 4) {
      int tm = w >> 1, tn = w & 1;
      v8f acc = {};
      for (int kk = 0; kk < DK_; kk += 32) {
        v16bf a  = load_a_vec(qg + (size_t)(l0 + 16 * tm) * DK_ + kk, DK_);
        v16bf bB = load_bt_vec(kg + (size_t)(l0 + 16 * tn) * DK_ + kk, DK_);
        acc = wmma_bf16(a, bB, acc);
      }
#pragma unroll
      for (int r = 0; r < 8; ++r) {
        int m = 16 * tm + r + hi8, nn = 16 * tn + cn;
        ab[m * 32 + nn] = f2bf((nn <= m) ? acc[r] : 0.f);
      }
    }
    __syncthreads();

    // Phase 6: o = q @ S + attn @ u2 -> delta[b, l, h, :]
    for (int it = 0; it < 4; ++it) {
      int tile = w * 4 + it, tm = tile >> 4, tj = tile & 15;
      v8f acc = {};
      for (int kk = 0; kk < DK_; kk += 32) {
        v16bf a  = load_a_vec(qg + (size_t)(l0 + 16 * tm) * DK_ + kk, DK_);
        v16bf bS = load_bt_vec(ST + (size_t)(16 * tj) * DK_ + kk, DK_);
        acc = wmma_bf16(a, bS, acc);
      }
      acc = wmma_bf16(load_a_vec(ab + 16 * tm * 32, 32),
                      load_bt_vec(ubT + (size_t)(16 * tj) * 32, 32), acc);
#pragma unroll
      for (int r = 0; r < 8; ++r) {
        int m = 16 * tm + r + hi8;
        delta[(((size_t)b * L_ + l0 + m) * H_ + h) * DV_ + 16 * tj + cn] =
            acc[r];
      }
    }
    __syncthreads();

    // Phase 7: S += k^T @ u2 (f32) + bf16 transposed mirror; tile-owned.
    for (int it = 0; it < 32; ++it) {
      int tile = w * 32 + it, ti = tile >> 4, tj = tile & 15;
      v16bf a  = load_a_vec(kTg + (size_t)(16 * ti) * L_ + l0, L_);
      v16bf bB = load_bt_vec(ubT + (size_t)(16 * tj) * 32, 32);
      float* Sp = S + (size_t)(16 * ti) * DV_ + 16 * tj;
      v8f cS = load_c_f32(Sp, DV_);
      cS = wmma_bf16(a, bB, cS);
      store_c_f32(Sp, DV_, cS);
      uint4 pk;
      unsigned short* ps = (unsigned short*)&pk;
#pragma unroll
      for (int r = 0; r < 8; ++r) ps[r] = f2bf(cS[r]);
      *(uint4*)(ST + (size_t)(16 * tj + cn) * DK_ + 16 * ti + hi8) = pk;
    }
    __syncthreads();
  }
}

// ---------------------------------------------------------------------------
// Per-head stats + router_in (bf16, K padded to 1056). One block per row.
// ---------------------------------------------------------------------------
__global__ __launch_bounds__(256) void stats_kernel(
    const float* __restrict__ x, const float* __restrict__ fs,
    const float* __restrict__ fl, const float* __restrict__ delta,
    unsigned short* __restrict__ rin) {
  __shared__ float red[256];
  const int row = blockIdx.x, t = threadIdx.x;
  unsigned short* rrow = rin + (size_t)row * RINP_;
  const float* xr = x + (size_t)row * D_;
  for (int j = 0; j < 4; ++j) rrow[t + 256 * j] = f2bf(xr[t + 256 * j]);
  if (t < RINP_ - RIN_) rrow[RIN_ + t] = 0;  // zero K-pad
  const float* srcs[3] = {fs, fl, delta};
  for (int s = 0; s < 3; ++s) {
    const float* src = srcs[s] + (size_t)row * D_;
    for (int h = 0; h < H_; ++h) {
      float v = src[h * 256 + t];
      float mean = block_reduce256(v, red) * (1.f / 256.f);
      float m2 = block_reduce256(v * v, red) * (1.f / 256.f);
      if (t == 0) {
        rrow[D_ + s * 8 + h] = f2bf(mean);
        rrow[D_ + s * 8 + 4 + h] = f2bf(sqrtf(fmaxf(m2 - mean * mean, 0.f)));
      }
      __syncthreads();
    }
  }
}

// ---------------------------------------------------------------------------
// Router head (N=12) + softmax/tau + mixture + per-head RMS norm -> bf16.
// ---------------------------------------------------------------------------
__global__ __launch_bounds__(256) void mix_kernel(
    const float* __restrict__ hmid, const float* __restrict__ Wr2,
    const float* __restrict__ br2, const float* __restrict__ ltg,
    const float* __restrict__ fs, const float* __restrict__ fl,
    const float* __restrict__ delta, const float* __restrict__ idsc,
    const float* __restrict__ vs, const float* __restrict__ onw,
    unsigned short* __restrict__ onorm) {
  __shared__ float red[256];
  __shared__ float lsh[12];
  __shared__ float psh[12];
  const int row = blockIdx.x, t = threadIdx.x;
  const float* hr = hmid + (size_t)row * RHID_;
  float part[12];
#pragma unroll
  for (int cc = 0; cc < 12; ++cc) part[cc] = 0.f;
  for (int jj = 0; jj < 8; ++jj) {
    int ix = t + 256 * jj;
    float hv = hr[ix];
#pragma unroll
    for (int cc = 0; cc < 12; ++cc) part[cc] += hv * Wr2[ix * 12 + cc];
  }
  for (int cc = 0; cc < 12; ++cc) {
    float s = block_reduce256(part[cc], red);
    if (t == 0) lsh[cc] = s + br2[cc];
    __syncthreads();
  }
  if (t < H_) {
    int h = t;
    float tau = expf(ltg[h >> 1]);  // GROUP=2; log_tau_head contributes 0
    float a = lsh[h * 3 + 0] / tau, bb = lsh[h * 3 + 1] / tau,
          cc = lsh[h * 3 + 2] / tau;
    float mx = fmaxf(a, fmaxf(bb, cc));
    float ea = expf(a - mx), eb = expf(bb - mx), ec = expf(cc - mx);
    float sum = ea + eb + ec, sc = (1.f - 3.f * R_EPS_);
    psh[h * 3 + 0] = ea / sum * sc + R_EPS_;
    psh[h * 3 + 1] = eb / sum * sc + R_EPS_;
    psh[h * 3 + 2] = ec / sum * sc + R_EPS_;
  }
  __syncthreads();
  for (int h = 0; h < H_; ++h) {
    int c = h * 256 + t;
    size_t ix = (size_t)row * D_ + c;
    float o = psh[h * 3 + 0] * fs[ix] + psh[h * 3 + 1] * fl[ix] +
              psh[h * 3 + 2] * delta[ix] + idsc[row * H_ + h] * vs[ix];
    float ms = block_reduce256(o * o, red) * (1.f / 256.f);
    onorm[ix] = f2bf(o * rsqrtf(ms + 1e-5f) * onw[t]);
  }
}

// ---------------------------------------------------------------------------
// Host-side orchestration
// ---------------------------------------------------------------------------
extern "C" void kernel_launch(void* const* d_in, const int* in_sizes, int n_in,
                              void* d_out, int out_size, void* d_ws,
                              size_t ws_size, hipStream_t stream) {
  (void)in_sizes; (void)n_in; (void)out_size; (void)ws_size;
  const float* x      = (const float*)d_in[0];
  const float* Wq     = (const float*)d_in[1];
  const float* Wk     = (const float*)d_in[2];
  const float* Wv     = (const float*)d_in[3];
  const float* Wb     = (const float*)d_in[4];
  const float* conv_q = (const float*)d_in[5];
  const float* conv_k = (const float*)d_in[6];
  const float* conv_v = (const float*)d_in[7];
  const float* fir_s  = (const float*)d_in[8];
  const float* fir_l  = (const float*)d_in[9];
  const float* alpha  = (const float*)d_in[10];
  const float* Wid    = (const float*)d_in[11];
  const float* bid    = (const float*)d_in[12];
  const float* Wr1    = (const float*)d_in[13];
  const float* br1    = (const float*)d_in[14];
  const float* Wr2    = (const float*)d_in[15];
  const float* br2    = (const float*)d_in[16];
  const float* ltg    = (const float*)d_in[17];
  /* d_in[18] log_tau_head unused (×0 in reference) */
  const float* onw    = (const float*)d_in[19];
  const float* Wo     = (const float*)d_in[20];
  float* out = (float*)d_out;

  char* ws = (char*)d_ws;
  size_t off = 0;
  auto alloc = [&](size_t bytes) {
    void* p = ws + off;
    off += (bytes + 255) & ~(size_t)255;
    return p;
  };
  const size_t F = (size_t)ROWS_ * D_;  // 8192*1024
  float* qc = (float*)alloc(F * 4);
  float* kc = (float*)alloc(F * 4);
  float* vc = (float*)alloc(F * 4);
  float* qs = (float*)alloc(F * 4);
  float* ks = (float*)alloc(F * 4);
  float* vs = (float*)alloc(F * 4);
  unsigned short* qn   = (unsigned short*)alloc(F * 2);
  unsigned short* kn   = (unsigned short*)alloc(F * 2);
  unsigned short* kTn  = (unsigned short*)alloc(F * 2);
  unsigned short* kbn  = (unsigned short*)alloc(F * 2);
  unsigned short* kbTn = (unsigned short*)alloc(F * 2);
  unsigned short* vbTn = (unsigned short*)alloc(F * 2);
  float* Sg = (float*)alloc((size_t)B_ * H_ * DK_ * DV_ * 4);
  unsigned short* STg = (unsigned short*)alloc((size_t)B_ * H_ * DK_ * DV_ * 2);
  float* deltab = (float*)alloc(F * 4);
  float* beta = (float*)alloc((size_t)ROWS_ * H_ * 4);
  float* idsc = (float*)alloc((size_t)ROWS_ * H_ * 4);
  unsigned short* rinb = (unsigned short*)alloc((size_t)ROWS_ * RINP_ * 2);
  float* hmid = (float*)alloc((size_t)ROWS_ * RHID_ * 4);
  unsigned short* xbf  = (unsigned short*)alloc(F * 2);
  unsigned short* WqT  = (unsigned short*)alloc((size_t)D_ * D_ * 2);
  unsigned short* WkT  = (unsigned short*)alloc((size_t)D_ * D_ * 2);
  unsigned short* WvT  = (unsigned short*)alloc((size_t)D_ * D_ * 2);
  unsigned short* WoT  = (unsigned short*)alloc((size_t)D_ * D_ * 2);
  unsigned short* Wr1T = (unsigned short*)alloc((size_t)RHID_ * RINP_ * 2);
  float* fs = qc;            // raw projections dead after conv: reuse
  float* fl = kc;
  unsigned short* onormb = xbf;  // x_bf dead after QKV GEMMs: reuse

  const dim3 blk(256);
  auto ggrid = [](int M, int N) { return dim3((N + 63) / 64, (M + 63) / 64); };
  auto eblk = [](size_t n) { return (int)((n + 255) / 256); };

  // 0) operand conversion (bf16 activations, transposed bf16 weights)
  cvt_bf16_kernel<<<eblk(F), blk, 0, stream>>>(x, xbf, (long)F);
  transpose_bf16_kernel<<<eblk((size_t)D_ * D_), blk, 0, stream>>>(Wq, WqT, D_, D_, D_);
  transpose_bf16_kernel<<<eblk((size_t)D_ * D_), blk, 0, stream>>>(Wk, WkT, D_, D_, D_);
  transpose_bf16_kernel<<<eblk((size_t)D_ * D_), blk, 0, stream>>>(Wv, WvT, D_, D_, D_);
  transpose_bf16_kernel<<<eblk((size_t)D_ * D_), blk, 0, stream>>>(Wo, WoT, D_, D_, D_);
  transpose_bf16_kernel<<<eblk((size_t)RHID_ * RINP_), blk, 0, stream>>>(
      Wr1, Wr1T, RIN_, RHID_, RINP_);
  // 1) QKV projections (bf16 WMMA, async-staged)
  gemm_bf16_kernel<<<ggrid(ROWS_, D_), blk, 0, stream>>>(xbf, WqT, nullptr, qc,
                                                         ROWS_, D_, D_, ACT_NONE);
  gemm_bf16_kernel<<<ggrid(ROWS_, D_), blk, 0, stream>>>(xbf, WkT, nullptr, kc,
                                                         ROWS_, D_, D_, ACT_NONE);
  gemm_bf16_kernel<<<ggrid(ROWS_, D_), blk, 0, stream>>>(xbf, WvT, nullptr, vc,
                                                         ROWS_, D_, D_, ACT_NONE);
  // 2) causal dwconv + SiLU
  dwconv_kernel<<<eblk(F), blk, 0, stream>>>(qc, conv_q, qs, 4, 1);
  dwconv_kernel<<<eblk(F), blk, 0, stream>>>(kc, conv_k, ks, 4, 1);
  dwconv_kernel<<<eblk(F), blk, 0, stream>>>(vc, conv_v, vs, 4, 1);
  // 3) gates
  beta_id_kernel<<<ROWS_, blk, 0, stream>>>(x, Wb, Wid, bid, alpha, beta, idsc);
  // 4) l2norm / beta-scale / operand layouts
  prep_kernel<<<ROWS_, blk, 0, stream>>>(qs, ks, vs, beta, qn, kn, kTn, kbn,
                                         kbTn, vbTn);
  // 5) chunkwise delta-rule scan
  delta_kernel<<<B_ * H_, blk, 0, stream>>>(qn, kn, kTn, kbn, kbTn, vbTn, Sg,
                                            STg, deltab);
  // 6) FIR branches
  dwconv_kernel<<<eblk(F), blk, 0, stream>>>(vs, fir_s, fs, 3, 0);
  dwconv_kernel<<<eblk(F), blk, 0, stream>>>(vs, fir_l, fl, 31, 0);
  // 7) stats + router input (bf16, padded K)
  stats_kernel<<<ROWS_, blk, 0, stream>>>(x, fs, fl, deltab, rinb);
  // 8) router layer 1 (bias + exact GELU epilogue)
  gemm_bf16_kernel<<<ggrid(ROWS_, RHID_), blk, 0, stream>>>(
      rinb, Wr1T, br1, hmid, ROWS_, RHID_, RINP_, ACT_GELU);
  // 9) logits + softmax + mixture + per-head RMS norm -> bf16
  mix_kernel<<<ROWS_, blk, 0, stream>>>(hmid, Wr2, br2, ltg, fs, fl, deltab,
                                        idsc, vs, onw, onormb);
  // 10) output projection
  gemm_bf16_kernel<<<ggrid(ROWS_, D_), blk, 0, stream>>>(
      onormb, WoT, nullptr, out, ROWS_, D_, D_, ACT_NONE);
}
